// ECCEdgeClassifier_28982439313930
// MI455X (gfx1250) — compile-verified, matrix operand order checked by reading, not run on hardware
//
#include <hip/hip_runtime.h>
#include <hip/hip_bf16.h>

#define N_NODES   65536
#define E_EDGES   262144
#define EP_PAIRS  131072
#define IN_DIM    16
#define H_DIM     32
#define EDGE_IN   8
#define KSIZE     3
#define DD        5
#define SS        32

typedef __attribute__((ext_vector_type(16))) _Float16 v16h;
typedef __attribute__((ext_vector_type(8)))  _Float16 h8;
typedef __attribute__((ext_vector_type(4)))  _Float16 h4;
typedef __attribute__((ext_vector_type(8)))  float    v8f;
typedef __attribute__((ext_vector_type(4)))  float    f4v;

// ---------------- workspace layout (bytes) ----------------
static constexpr size_t OFF_MM  = 0;                                        // 64 B (min enc [0..4], max enc [8..12])
static constexpr size_t OFF_DEG = 256;                                      // N*4
static constexpr size_t OFF_AGG = OFF_DEG + (size_t)N_NODES * 4;            // N*32*4
static constexpr size_t OFF_ZA  = OFF_AGG + (size_t)N_NODES * H_DIM * 4;    // N*32*4
static constexpr size_t OFF_ZB  = OFF_ZA  + (size_t)N_NODES * H_DIM * 4;    // N*32*4
static constexpr size_t OFF_BAS = OFF_ZB  + (size_t)N_NODES * H_DIM * 4;    // E*32*4
static constexpr size_t OFF_WI  = OFF_BAS + (size_t)E_EDGES * SS * 4;       // E*32 (u8)
static constexpr size_t OFF_W1F = OFF_WI  + (size_t)E_EDGES * SS;           // 6*32*16 halves (fragment order)
static constexpr size_t OFF_W2F = OFF_W1F + 6 * 32 * 16 * 2;                // 32*16 halves (fragment order)

// ---------------- helpers ----------------
__device__ __forceinline__ unsigned enc_f(float f) {
  unsigned u = __float_as_uint(f);
  return ((int)u < 0) ? ~u : (u | 0x80000000u);
}
__device__ __forceinline__ float dec_f(unsigned u) {
  return ((int)u < 0) ? __uint_as_float(u & 0x7FFFFFFFu) : __uint_as_float(~u);
}
// nan_to_num(nan=0, +inf=1, -inf=0) then col-4 remap, as in reference
__device__ __forceinline__ float fix_val(float v, int d) {
  unsigned b = __float_as_uint(v);
  if ((b & 0x7F800000u) == 0x7F800000u)
    v = (b & 0x007FFFFFu) ? 0.f : ((b >> 31) ? 0.f : 1.f);
  if (d == 4) v = fminf(fmaxf((v + 1.f) * 0.5f, 0.f), 1.f);
  return v;
}

// ---------------- per-column min/max of processed pseudo coords ----------------
__global__ void k_minmax(const float* __restrict__ ea, unsigned* __restrict__ mm) {
  int tid = blockIdx.x * blockDim.x + threadIdx.x;
  int stride = gridDim.x * blockDim.x;
  unsigned mn[DD], mx[DD];
  #pragma unroll
  for (int d = 0; d < DD; ++d) { mn[d] = 0xFFFFFFFFu; mx[d] = 0u; }
  for (int e = tid; e < E_EDGES; e += stride) {
    #pragma unroll
    for (int d = 0; d < DD; ++d) {
      unsigned u = enc_f(fix_val(ea[(size_t)e * EDGE_IN + d], d));
      mn[d] = (u < mn[d]) ? u : mn[d];
      mx[d] = (u > mx[d]) ? u : mx[d];
    }
  }
  #pragma unroll
  for (int m = 16; m >= 1; m >>= 1) {
    #pragma unroll
    for (int d = 0; d < DD; ++d) {
      unsigned a = (unsigned)__shfl_xor((int)mn[d], m, 32);
      unsigned b = (unsigned)__shfl_xor((int)mx[d], m, 32);
      mn[d] = (a < mn[d]) ? a : mn[d];
      mx[d] = (b > mx[d]) ? b : mx[d];
    }
  }
  if ((threadIdx.x & 31) == 0) {
    #pragma unroll
    for (int d = 0; d < DD; ++d) {
      atomicMin(&mm[d], mn[d]);
      atomicMax(&mm[8 + d], mx[d]);
    }
  }
}

// ---------------- spline basis + weight indices (per edge) ----------------
__global__ void k_basis(const float* __restrict__ ea, const unsigned* __restrict__ mm,
                        float* __restrict__ basis, unsigned char* __restrict__ wi) {
  int e = blockIdx.x * blockDim.x + threadIdx.x;
  if (e >= E_EDGES) return;
  float frac[DD];
  int fi[DD];
  #pragma unroll
  for (int d = 0; d < DD; ++d) {
    float v = fix_val(ea[(size_t)e * EDGE_IN + d], d);
    float cmin = dec_f(mm[d]), cmax = dec_f(mm[8 + d]);
    float den = fmaxf(cmax - cmin, 1e-6f);
    float ps = fminf(fmaxf((v - cmin) / den, 0.f), 1.f);
    float p = ps * (float)(KSIZE - 1);
    float f = fminf(fmaxf(floorf(p), 0.f), (float)(KSIZE - 2));
    frac[d] = p - f;
    fi[d] = (int)f;
  }
  const int strides[DD] = {1, 3, 9, 27, 81};
  for (int s = 0; s < SS; ++s) {
    float b = 1.f;
    int w = 0;
    #pragma unroll
    for (int d = 0; d < DD; ++d) {
      int bit = (s >> d) & 1;
      b *= bit ? frac[d] : (1.f - frac[d]);
      w += (fi[d] + bit) * strides[d];
    }
    basis[(size_t)e * SS + s] = b;
    wi[(size_t)e * SS + s] = (unsigned char)w;
  }
}

// ---------------- spline-conv messages + scatter-aggregate ----------------
// one wave per edge; lane = output channel; x_j broadcasts hoisted out of the tap loop
__global__ void k_msg(const float* __restrict__ z, int in_dim,
                      const float* __restrict__ W,
                      const float* __restrict__ basis, const unsigned char* __restrict__ wi,
                      const int* __restrict__ ei,
                      float* __restrict__ agg, float* __restrict__ deg) {
  int lane = threadIdx.x & 31;
  int e = blockIdx.x * (blockDim.x >> 5) + (threadIdx.x >> 5);
  if (e >= E_EDGES) return;
  int src = ei[e];
  int dst = ei[E_EDGES + e];
  float xj = (lane < in_dim) ? z[(size_t)src * in_dim + lane] : 0.f;
  float xv[H_DIM];
  #pragma unroll
  for (int i = 0; i < H_DIM; ++i) xv[i] = __shfl(xj, i, 32);
  float bs = basis[(size_t)e * SS + lane];
  int   wv = (int)wi[(size_t)e * SS + lane];
  float acc = 0.f;
  for (int s = 0; s < SS; ++s) {
    float b  = __shfl(bs, s, 32);
    int   wd = __shfl(wv, s, 32);
    const float* Wp = W + ((size_t)wd * in_dim) * H_DIM + lane;
    if (s + 1 < SS) {
      int wn = __shfl(wv, s + 1, 32);
      __builtin_prefetch(W + ((size_t)wn * in_dim) * H_DIM + lane, 0, 1);
    }
    float t = 0.f;
    if (in_dim == IN_DIM) {
      #pragma unroll
      for (int i = 0; i < IN_DIM; ++i) t = fmaf(xv[i], Wp[(size_t)i * H_DIM], t);
    } else {
      #pragma unroll
      for (int i = 0; i < H_DIM; ++i) t = fmaf(xv[i], Wp[(size_t)i * H_DIM], t);
    }
    acc = fmaf(b, t, acc);
  }
  atomicAdd(&agg[(size_t)dst * H_DIM + lane], acc);
  if (lane == 0) atomicAdd(&deg[dst], 1.f);
}

// ---------------- node update: deg-norm + root linear + LN + ReLU (+resid) ----------------
__global__ void k_node(const float* __restrict__ agg, const float* __restrict__ deg,
                       const float* __restrict__ zin, int in_dim,
                       const float* __restrict__ root, const float* __restrict__ cb,
                       const float* __restrict__ g, const float* __restrict__ be,
                       int resid, float* __restrict__ zout) {
  int lane = threadIdx.x & 31;
  int n = blockIdx.x * (blockDim.x >> 5) + (threadIdx.x >> 5);
  if (n >= N_NODES) return;
  float zi = (lane < in_dim) ? zin[(size_t)n * in_dim + lane] : 0.f;
  float zv[H_DIM];
  #pragma unroll
  for (int i = 0; i < H_DIM; ++i) zv[i] = __shfl(zi, i, 32);
  float r = 0.f;
  for (int i = 0; i < in_dim; ++i) r = fmaf(zv[i], root[(size_t)i * H_DIM + lane], r);
  float dg = fmaxf(deg[n], 1.f);
  float val = agg[(size_t)n * H_DIM + lane] / dg + r + cb[lane];
  float s = val, s2 = val * val;
  #pragma unroll
  for (int m = 16; m >= 1; m >>= 1) {
    s  += __shfl_xor(s, m, 32);
    s2 += __shfl_xor(s2, m, 32);
  }
  float mu  = s * (1.f / 32.f);
  float var = s2 * (1.f / 32.f) - mu * mu;
  float ln = (val - mu) * rsqrtf(var + 1e-5f) * g[lane] + be[lane];
  float o = fmaxf(ln, 0.f);
  if (resid) o += zi;
  zout[(size_t)n * H_DIM + lane] = o;
}

// ---------------- convert MLP weights to f16, pre-swizzled into WMMA B-fragment order ----
// w1frag[(c*2+ct)*32 + lane][e] , c=K-chunk(3), ct=col-tile(2)  (mw1 padded K 72->96)
// w2frag[lane][e]
__global__ void k_wcvt(const float* __restrict__ mw1, const float* __restrict__ mw2,
                       _Float16* __restrict__ w1f, _Float16* __restrict__ w2f) {
  int t = threadIdx.x;
  for (int f = t; f < 6 * 32; f += blockDim.x) {
    int c    = f >> 6;          // 0..2
    int ct   = (f >> 5) & 1;    // 0..1
    int lane = f & 31;
    int hh = lane >> 4;
    int n  = ct * 16 + (lane & 15);
    #pragma unroll
    for (int e = 0; e < 16; ++e) {
      int k = c * 32 + ((e & 8) ? 16 : 0) + hh * 8 + (e & 7);
      w1f[(size_t)f * 16 + e] = (_Float16)(k < 72 ? mw1[k * 32 + n] : 0.f);
    }
  }
  for (int lane = t; lane < 32; lane += blockDim.x) {
    int hh = lane >> 4;
    int n  = lane & 15;
    #pragma unroll
    for (int e = 0; e < 16; ++e) {
      int k = ((e & 8) ? 16 : 0) + hh * 8 + (e & 7);
      w2f[(size_t)lane * 16 + e] = (_Float16)mw2[k * 16 + n];
    }
  }
}

// ---------------- edge-pair MLP via f16 WMMA: 16 pairs per wave ----------------
__global__ void __launch_bounds__(256) k_mlp(const float* __restrict__ z,
                                             const int* __restrict__ ep,
                                             const float* __restrict__ eap,
                                             const _Float16* __restrict__ w1f,
                                             const _Float16* __restrict__ w2f,
                                             const float* __restrict__ mb1,
                                             const float* __restrict__ mb2,
                                             const float* __restrict__ mw3,
                                             const float* __restrict__ mb3,
                                             float* __restrict__ out) {
  // 16-byte aligned so ds_load_b128 / 8-byte ds_store are legal
  __shared__ __attribute__((aligned(16))) _Float16 sfeats[8][16][96]; // [z[u]|z[v]|ea|0-pad]
  __shared__ __attribute__((aligned(16))) _Float16 sh1[8][16][32];
  int lane = threadIdx.x & 31;
  int wid  = threadIdx.x >> 5;
  int tile = blockIdx.x * 8 + wid;
  int base = tile * 16;

  // stage features: 16 rows x 24 float4 groups (cols 72..95 zero-padded)
  for (int t = lane; t < 16 * 24; t += 32) {
    int row = t / 24;
    int col = (t % 24) * 4;
    int pe = base + row;
    f4v v;
    if (col < 32) {
      v = *(const f4v*)&z[(size_t)ep[pe] * H_DIM + col];
    } else if (col < 64) {
      v = *(const f4v*)&z[(size_t)ep[EP_PAIRS + pe] * H_DIM + (col - 32)];
    } else if (col < 72) {
      v = *(const f4v*)&eap[(size_t)pe * EDGE_IN + (col - 64)];
    } else {
      v = (f4v){0.f, 0.f, 0.f, 0.f};
    }
    h4 hv;
    #pragma unroll
    for (int j = 0; j < 4; ++j) hv[j] = (_Float16)v[j];
    *(h4*)&sfeats[wid][row][col] = hv;
  }
  __syncthreads();

  int m  = lane & 15;
  int hh = lane >> 4;
  const h8* arow = (const h8*)&sfeats[wid][m][0]; // 12 x h8 per row

  // layer 1: 16x96 @ 96x32 -> 16x32 (3 K-chunks, 2 col tiles), B from pre-swizzled frags
  v8f acc0 = {};
  v8f acc1 = {};
  #pragma unroll
  for (int c = 0; c < 3; ++c) {
    h8 lo = arow[c * 4 + hh];
    h8 hi = arow[c * 4 + hh + 2];
    v16h a = __builtin_shufflevector(lo, hi, 0, 1, 2, 3, 4, 5, 6, 7,
                                             8, 9, 10, 11, 12, 13, 14, 15);
    v16h b0 = *(const v16h*)&w1f[((size_t)(c * 2 + 0) * 32 + lane) * 16];
    v16h b1 = *(const v16h*)&w1f[((size_t)(c * 2 + 1) * 32 + lane) * 16];
    acc0 = __builtin_amdgcn_wmma_f32_16x16x32_f16(false, a, false, b0, (short)0, acc0, false, false);
    acc1 = __builtin_amdgcn_wmma_f32_16x16x32_f16(false, a, false, b1, (short)0, acc1, false, false);
  }
  #pragma unroll
  for (int r = 0; r < 8; ++r) {
    int M = r + 8 * hh;
    sh1[wid][M][m]      = (_Float16)fmaxf(acc0[r] + mb1[m], 0.f);
    sh1[wid][M][16 + m] = (_Float16)fmaxf(acc1[r] + mb1[16 + m], 0.f);
  }
  __syncthreads();

  // layer 2: 16x32 @ 32x16 -> 16x16
  const h8* hrow = (const h8*)&sh1[wid][m][0]; // 4 x h8 per row
  h8 lo2 = hrow[hh];
  h8 hi2 = hrow[hh + 2];
  v16h a2 = __builtin_shufflevector(lo2, hi2, 0, 1, 2, 3, 4, 5, 6, 7,
                                              8, 9, 10, 11, 12, 13, 14, 15);
  v16h b2 = *(const v16h*)&w2f[(size_t)lane * 16];
  v8f acc2 = {};
  acc2 = __builtin_amdgcn_wmma_f32_16x16x32_f16(false, a2, false, b2, (short)0, acc2, false, false);

  // layer 3: dot with mw3 (16) + reduce across the 16 columns
  float w3 = mw3[m];
  float b3 = mb3[0];
  #pragma unroll
  for (int r = 0; r < 8; ++r) {
    float h2 = fmaxf(acc2[r] + mb2[m], 0.f);
    float p = h2 * w3;
    #pragma unroll
    for (int msk = 1; msk < 16; msk <<= 1) p += __shfl_xor(p, msk, 32);
    if (m == 0) out[base + r + 8 * hh] = p + b3;
  }
}

// ---------------- host launcher ----------------
extern "C" void kernel_launch(void* const* d_in, const int* in_sizes, int n_in,
                              void* d_out, int out_size, void* d_ws, size_t ws_size,
                              hipStream_t stream) {
  const float* x     = (const float*)d_in[0];
  const int*   gei   = (const int*)  d_in[1];
  const float* gea   = (const float*)d_in[2];
  const int*   ep    = (const int*)  d_in[3];
  const float* eap   = (const float*)d_in[4];
  const float* W0    = (const float*)d_in[5];
  const float* W1    = (const float*)d_in[6];
  const float* W2    = (const float*)d_in[7];
  const float* root0 = (const float*)d_in[8];
  const float* root1 = (const float*)d_in[9];
  const float* root2 = (const float*)d_in[10];
  const float* cb0   = (const float*)d_in[11];
  const float* cb1   = (const float*)d_in[12];
  const float* cb2   = (const float*)d_in[13];
  const float* g0    = (const float*)d_in[14];
  const float* g1    = (const float*)d_in[15];
  const float* g2    = (const float*)d_in[16];
  const float* be0   = (const float*)d_in[17];
  const float* be1   = (const float*)d_in[18];
  const float* be2   = (const float*)d_in[19];
  const float* mw1   = (const float*)d_in[20];
  const float* mb1   = (const float*)d_in[21];
  const float* mw2   = (const float*)d_in[22];
  const float* mb2   = (const float*)d_in[23];
  const float* mw3   = (const float*)d_in[24];
  const float* mb3   = (const float*)d_in[25];

  char* ws = (char*)d_ws;
  unsigned*      mm    = (unsigned*)     (ws + OFF_MM);
  float*         deg   = (float*)        (ws + OFF_DEG);
  float*         agg   = (float*)        (ws + OFF_AGG);
  float*         zA    = (float*)        (ws + OFF_ZA);
  float*         zB    = (float*)        (ws + OFF_ZB);
  float*         basis = (float*)        (ws + OFF_BAS);
  unsigned char* wib   = (unsigned char*)(ws + OFF_WI);
  _Float16*      w1f   = (_Float16*)     (ws + OFF_W1F);
  _Float16*      w2f   = (_Float16*)     (ws + OFF_W2F);

  const size_t degagg_bytes = (size_t)N_NODES * 4 + (size_t)N_NODES * H_DIM * 4;

  // pseudo normalization stats
  hipMemsetAsync(ws + OFF_MM, 0xFF, 20, stream);      // min enc init = +inf
  hipMemsetAsync(ws + OFF_MM + 32, 0x00, 20, stream); // max enc init = -inf
  k_minmax<<<512, 256, 0, stream>>>(gea, mm);
  k_basis<<<E_EDGES / 256, 256, 0, stream>>>(gea, mm, basis, wib);

  // layer 0: x(16) -> zA(32)
  hipMemsetAsync(deg, 0, degagg_bytes, stream);
  k_msg<<<E_EDGES / 8, 256, 0, stream>>>(x, IN_DIM, W0, basis, wib, gei, agg, deg);
  k_node<<<N_NODES / 8, 256, 0, stream>>>(agg, deg, x, IN_DIM, root0, cb0, g0, be0, 0, zA);

  // layer 1: zA -> zB (+resid)
  hipMemsetAsync(deg, 0, degagg_bytes, stream);
  k_msg<<<E_EDGES / 8, 256, 0, stream>>>(zA, H_DIM, W1, basis, wib, gei, agg, deg);
  k_node<<<N_NODES / 8, 256, 0, stream>>>(agg, deg, zA, H_DIM, root1, cb1, g1, be1, 1, zB);

  // layer 2: zB -> zA (+resid)
  hipMemsetAsync(deg, 0, degagg_bytes, stream);
  k_msg<<<E_EDGES / 8, 256, 0, stream>>>(zB, H_DIM, W2, basis, wib, gei, agg, deg);
  k_node<<<N_NODES / 8, 256, 0, stream>>>(agg, deg, zB, H_DIM, root2, cb2, g2, be2, 1, zA);

  // edge-pair MLP (WMMA f16, fragment-order weights)
  k_wcvt<<<1, 256, 0, stream>>>(mw1, mw2, w1f, w2f);
  k_mlp<<<EP_PAIRS / 16 / 8, 256, 0, stream>>>(zA, ep, eap, w1f, w2f, mb1, mb2, mw3, mb3,
                                               (float*)d_out);
}
